// CoordUpdate_901943132401
// MI455X (gfx1250) — compile-verified, matrix operand order checked by reading, not run on hardware
//
#include <hip/hip_runtime.h>
#include <hip/hip_bf16.h>

typedef __attribute__((ext_vector_type(16))) __bf16 v16bf;
typedef __attribute__((ext_vector_type(8)))  float  v8f;

#define E_TOTAL  800000
#define NNODES   50000
#define HID      128
#define DIN      258
#define TILE_M   128   // edges per block (8 waves x 16)

union Frag {
    v16bf v;
    uint4 q[2];
};

// K-offset pattern inside a 32-wide K block for 16-bit A/B fragments
// (ISA 05_wmma): lanes 0-15 hold K {0..7,16..23}, lanes 16-31 hold {8..15,24..31}.
__device__ __forceinline__ int kpat(int v, int kh) {
    return ((v < 4) ? (2 * v) : (16 + 2 * (v - 4))) + 8 * kh;
}

// f32 -> bf16 RNE (used in the weight prepass and the layer-1 SiLU store)
__device__ __forceinline__ __bf16 f2bf_rne(float f) {
    unsigned int u = __builtin_bit_cast(unsigned int, f);
    unsigned int r = (u + 0x7FFFu + ((u >> 16) & 1u)) >> 16;
    return __builtin_bit_cast(__bf16, (unsigned short)(r & 0xFFFFu));
}

// pack two f32 -> bf16x2 dword: 2 adds + 1 v_perm_b32
__device__ __forceinline__ unsigned int pack2bf(float a, float b) {
    unsigned int ua = __builtin_bit_cast(unsigned int, a) + 0x8000u;
    unsigned int ub = __builtin_bit_cast(unsigned int, b) + 0x8000u;
    return __builtin_amdgcn_perm(ub, ua, 0x07060302u);  // {bf(b), bf(a)}
}

// fast silu: x * rcp(1+exp(-x)) -> v_exp_f32 + v_rcp_f32 (bf16-accurate)
__device__ __forceinline__ float silu(float x) {
    return x * __builtin_amdgcn_rcpf(1.0f + __expf(-x));
}

// fast tanh: 1 - 2/(exp(2x)+1); correct saturation at +/-inf
__device__ __forceinline__ float tanh_fast(float x) {
    return 1.0f - 2.0f * __builtin_amdgcn_rcpf(__expf(2.0f * x) + 1.0f);
}

// ---------------- prepass: W1/W2 -> bf16 in B-fragment order ----------------
// ws layout: [64 frags W1 | 32 frags W2], each frag = 32 lanes x 16 bf16.
// Frag id = kb*8 + nt.
__global__ void conv_weights_kernel(const float* __restrict__ W1,
                                    const float* __restrict__ W2,
                                    __bf16* __restrict__ ws) {
    int gid = blockIdx.x * blockDim.x + threadIdx.x;
    if (gid >= 3072) return;
    const float* W;
    int ld, frag, lane;
    __bf16* out;
    if (gid < 2048) {                 // W1: 64 frags
        W = W1; ld = DIN;
        frag = gid >> 5; lane = gid & 31;
        out = ws + (size_t)gid * 16;
    } else {                          // W2: 32 frags
        W = W2; ld = HID;
        int g2 = gid - 2048;
        frag = g2 >> 5; lane = g2 & 31;
        out = ws + (size_t)2048 * 16 + (size_t)g2 * 16;
    }
    int kb = frag >> 3, nt = frag & 7;
    int n  = nt * 16 + (lane & 15);
    int kh = lane >> 4;
    const float* rowp = W + (size_t)n * ld + kb * 32;
#pragma unroll
    for (int v = 0; v < 8; ++v) {
        int kk = kpat(v, kh);
        out[2 * v]     = f2bf_rne(rowp[kk]);
        out[2 * v + 1] = f2bf_rne(rowp[kk + 1]);
    }
}

// B fragment: two global_load_b128 from pre-swizzled ws
__device__ __forceinline__ v16bf load_b_ws(const __bf16* __restrict__ ws,
                                           int frag, int lane) {
    Frag f;
    const uint4* p = (const uint4*)(ws + ((size_t)frag * 32 + lane) * 16);
    f.q[0] = p[0];
    f.q[1] = p[1];
    return f.v;
}

// A fragment: two ds_load_b128 from pre-swizzled LDS
__device__ __forceinline__ v16bf load_a_lds(const __bf16* base, int frag,
                                            int lane) {
    Frag f;
    const uint4* p = (const uint4*)(base + ((size_t)frag * 32 + lane) * 16);
    f.q[0] = p[0];
    f.q[1] = p[1];
    return f.v;
}

__global__ void init_out_kernel(const float* __restrict__ coord,
                                float* __restrict__ out, int n) {
    int i = blockIdx.x * blockDim.x + threadIdx.x;
    if (i < n) out[i] = coord[i];
}

__global__ __launch_bounds__(256)
void egnn_edge_kernel(const float* __restrict__ h,
                      const int*   __restrict__ edge_index,
                      const float* __restrict__ coord_diff,
                      const float* __restrict__ edge_attr,
                      const float* __restrict__ W1, const float* __restrict__ b1,
                      const float* __restrict__ b2, const float* __restrict__ W3,
                      const __bf16* __restrict__ wsW1,
                      const __bf16* __restrict__ wsW2,
                      float* __restrict__ out) {
    // A operands in fragment order: [(wave,kb)][lane][16]
    __shared__ __bf16 sXf[8 * 8 * 32 * 16];   // layer-1 A: 8 waves x 8 kb  (64 KB)
    __shared__ __bf16 sH1f[8 * 4 * 32 * 16];  // layer-2 A: 8 waves x 4 kb  (32 KB)
    __shared__ float  sB1[HID], sB2[HID], sW3[HID], sWa[HID], sWb[HID];
    __shared__ float  sEA0[TILE_M], sEA1[TILE_M], sS[TILE_M];
    __shared__ int    sRow[TILE_M], sCol[TILE_M];

    const int tid = threadIdx.x;
    const int e0  = blockIdx.x * TILE_M;

    if (tid < HID) {
        sB1[tid] = b1[tid];
        sB2[tid] = b2[tid];
        sW3[tid] = W3[tid];
        sWa[tid] = W1[(size_t)tid * DIN + 256];  // edge_attr columns of W1
        sWb[tid] = W1[(size_t)tid * DIN + 257];
    }
    if (tid < TILE_M) {
        int e = e0 + tid;
        if (e >= E_TOTAL) e = E_TOTAL - 1;       // E = 6250*128 exactly; safety
        sRow[tid] = edge_index[e];
        sCol[tid] = edge_index[E_TOTAL + e];
        sEA0[tid] = edge_attr[2 * e];
        sEA1[tid] = edge_attr[2 * e + 1];
    }
    __syncthreads();

    // Gather h[row]|h[col] -> bf16, written directly in A-fragment order.
    // Each q handles one float4 = 4 consecutive K of one edge -> one 8B LDS store.
    for (int q = tid; q < TILE_M * 64; q += 256) {
        int m    = q >> 6;
        int part = q & 63;                        // 0..31 row half, 32..63 col half
        int node = (part < 32) ? sRow[m] : sCol[m];
        int c4   = part & 31;
        float4 vv = ((const float4*)(h + (size_t)node * HID))[c4];
        int k   = ((part < 32) ? 0 : HID) + c4 * 4;   // global K 0..255
        int w   = m >> 4;
        int kb  = k >> 5;
        int kin = k & 31;
        int g   = kin >> 3;
        int kh  = g & 1;
        int i0  = kin - 8 * ((g + 1) >> 1);           // element index within frag
        int lane_dst = (m & 15) + 16 * kh;
        unsigned int d0 = pack2bf(vv.x, vv.y);
        unsigned int d1 = pack2bf(vv.z, vv.w);
        uint2* dst = (uint2*)&sXf[(((w * 8 + kb) * 32 + lane_dst) * 16) + i0];
        *dst = make_uint2(d0, d1);
    }
    __syncthreads();

    const int wave = tid >> 5, lane = tid & 31;
    const int m0 = wave * 16;
    const int ln = lane & 15, lh = lane >> 4;

    // ---------------- layer 1: [16 x 258] @ [258 x 128] ----------------
    // C layout: VGPR v, lane l -> M = m0 + v + 8*(l/16), N = nt*16 + l%16.
    v8f acc[8];
#pragma unroll
    for (int nt = 0; nt < 8; ++nt) {
        int n = nt * 16 + ln;
        float bn = sB1[n], wa = sWa[n], wb = sWb[n];
#pragma unroll
        for (int v = 0; v < 8; ++v) {
            int m = m0 + v + 8 * lh;
            acc[nt][v] = bn + sEA0[m] * wa + sEA1[m] * wb;  // bias + edge_attr tail
        }
    }
    for (int kb = 0; kb < 8; ++kb) {
        v16bf a = load_a_lds(sXf, wave * 8 + kb, lane);
#pragma unroll
        for (int nt = 0; nt < 8; ++nt) {
            v16bf b = load_b_ws(wsW1, kb * 8 + nt, lane);
            acc[nt] = __builtin_amdgcn_wmma_f32_16x16x32_bf16(
                false, a, false, b, (short)0, acc[nt], false, false);
        }
    }
    // SiLU -> bf16 -> LDS in layer-2 A-fragment order
#pragma unroll
    for (int nt = 0; nt < 8; ++nt) {
#pragma unroll
        for (int v = 0; v < 8; ++v) {
            int mloc = v + 8 * lh;                // M within this wave's tile
            int k    = nt * 16 + ln;              // K for layer 2 = N of layer 1
            int kb2  = k >> 5;
            int kin  = k & 31;
            int g    = kin >> 3;
            int kh2  = g & 1;
            int i    = kin - 8 * ((g + 1) >> 1);
            int lane_dst = mloc + 16 * kh2;
            sH1f[((wave * 4 + kb2) * 32 + lane_dst) * 16 + i] =
                f2bf_rne(silu(acc[nt][v]));
        }
    }
    __syncthreads();

    // ---------------- layer 2: [16 x 128] @ [128 x 128] ----------------
    v8f acc2[8];
#pragma unroll
    for (int nt = 0; nt < 8; ++nt) {
        float bn = sB2[nt * 16 + ln];
#pragma unroll
        for (int v = 0; v < 8; ++v) acc2[nt][v] = bn;
    }
    for (int kb = 0; kb < 4; ++kb) {
        v16bf a = load_a_lds(sH1f, wave * 4 + kb, lane);
#pragma unroll
        for (int nt = 0; nt < 8; ++nt) {
            v16bf b = load_b_ws(wsW2, kb * 8 + nt, lane);
            acc2[nt] = __builtin_amdgcn_wmma_f32_16x16x32_bf16(
                false, a, false, b, (short)0, acc2[nt], false, false);
        }
    }

    // ------------- s = silu(h2) . W3, butterfly over 16 lanes -------------
    float p[8];
#pragma unroll
    for (int v = 0; v < 8; ++v) p[v] = 0.0f;
#pragma unroll
    for (int nt = 0; nt < 8; ++nt) {
        float w3 = sW3[nt * 16 + ln];
#pragma unroll
        for (int v = 0; v < 8; ++v) p[v] += silu(acc2[nt][v]) * w3;
    }
#pragma unroll
    for (int v = 0; v < 8; ++v) {
#pragma unroll
        for (int off = 8; off >= 1; off >>= 1)
            p[v] += __shfl_xor(p[v], off, 16);   // wave32 half-wave reduce
    }
    if (ln == 0) {
#pragma unroll
        for (int v = 0; v < 8; ++v) sS[m0 + v + 8 * lh] = p[v];
    }
    __syncthreads();

    // ------------- tanh, scale, scatter-add into out -------------
    if (tid < TILE_M) {
        int e = e0 + tid;
        if (e < E_TOTAL) {
            float t = tanh_fast(sS[tid]) * (15.0f / 100.0f);  // COORDS_RANGE / NORM
            int r = sRow[tid];
            atomicAdd(&out[r * 3 + 0], coord_diff[3 * e + 0] * t);
            atomicAdd(&out[r * 3 + 1], coord_diff[3 * e + 1] * t);
            atomicAdd(&out[r * 3 + 2], coord_diff[3 * e + 2] * t);
        }
    }
}

extern "C" void kernel_launch(void* const* d_in, const int* in_sizes, int n_in,
                              void* d_out, int out_size, void* d_ws, size_t ws_size,
                              hipStream_t stream) {
    const float* h          = (const float*)d_in[0];
    const float* coord      = (const float*)d_in[1];
    const int*   edge_index = (const int*)d_in[2];
    const float* coord_diff = (const float*)d_in[3];
    const float* edge_attr  = (const float*)d_in[4];
    const float* W1         = (const float*)d_in[5];
    const float* b1         = (const float*)d_in[6];
    const float* W2         = (const float*)d_in[7];
    const float* b2         = (const float*)d_in[8];
    const float* W3         = (const float*)d_in[9];
    float* out  = (float*)d_out;
    __bf16* wsb = (__bf16*)d_ws;   // 96 KB used: 64KB W1 frags + 32KB W2 frags

    conv_weights_kernel<<<(3072 + 255) / 256, 256, 0, stream>>>(W1, W2, wsb);

    int ncoord = NNODES * 3;
    init_out_kernel<<<(ncoord + 255) / 256, 256, 0, stream>>>(coord, out, ncoord);

    int nblocks = (E_TOTAL + TILE_M - 1) / TILE_M;  // 6250
    egnn_edge_kernel<<<nblocks, 256, 0, stream>>>(
        h, edge_index, coord_diff, edge_attr, W1, b1, b2, W3,
        wsb, wsb + 2048 * 16, out);
}
